// GroupedQueryAttention_89816356094510
// MI455X (gfx1250) — matrix operand order repaired and checked by reading, not compiled, not run on hardware
//
#include <hip/hip_runtime.h>
#include <hip/hip_bf16.h>

// ---------------- Problem constants ----------------
#define SEQ     2048
#define D_IN    2048
#define D_OUT   2048
#define N_HEADS 32
#define N_KV    8
#define HEAD_DIM 64
#define KV_DIM  (N_KV * HEAD_DIM)   // 512

typedef __bf16 bf16_t;
typedef __attribute__((ext_vector_type(16))) __bf16    v16bf;
typedef __attribute__((ext_vector_type(8)))  float     v8f;
typedef __attribute__((ext_vector_type(4)))  unsigned  u32x4;
typedef __attribute__((ext_vector_type(8)))  int       i32x8;
typedef __attribute__((ext_vector_type(4)))  int       i32x4;

union AF { v16bf v; unsigned u[8]; u32x4 q[2]; };

#if defined(__has_builtin)
#if __has_builtin(__builtin_amdgcn_tensor_load_to_lds)
#define HAS_TDM 1
#endif
#endif
#ifndef HAS_TDM
#define HAS_TDM 0
#endif

__device__ __forceinline__ bf16_t f2bf(float f) {
    unsigned u = __builtin_bit_cast(unsigned, f);
    u += 0x7fffu + ((u >> 16) & 1u);            // round-to-nearest-even
    unsigned short h = (unsigned short)(u >> 16);
    return __builtin_bit_cast(bf16_t, h);
}

__device__ __forceinline__ v8f wmma_bf16(const v16bf& a, const v16bf& b, v8f c) {
    return __builtin_amdgcn_wmma_f32_16x16x32_bf16(false, a, false, b, (short)0, c,
                                                   false, false);
}

// 32-bit LDS byte offset of a generic pointer into shared memory
__device__ __forceinline__ unsigned lds_addr(const void* p) {
    return (unsigned)(size_t)(__attribute__((address_space(3))) const void*)p;
}

// Async DMA: 16 bytes global -> LDS, tracked by ASYNCcnt (CDNA5)
__device__ __forceinline__ void async_ld16(unsigned ldsoff, const bf16_t* gptr) {
    asm volatile("global_load_async_to_lds_b128 %0, %1, off"
                 :: "v"(ldsoff), "v"(gptr) : "memory");
}
__device__ __forceinline__ void wait_async0() {
    asm volatile("s_wait_asynccnt 0" ::: "memory");
}

// Tensor Data Mover: 2-D tile (tileK elems of 2B) x tileRows, row stride = strideElems.
// D# layout per CDNA5 ISA 8.3/8.4 (count=1, type=2=image, data_size=2B, no pad/iterate).
// This toolchain exposes the 6-arg builtin: (g0, g1, g2, g3, g4, cpol).
__device__ __forceinline__ void tdm_load_2d(unsigned lds_byte, const void* gaddr,
                                            unsigned tileK, unsigned tileRows,
                                            unsigned strideElems) {
#if HAS_TDM
    unsigned long long ga = (unsigned long long)gaddr;
    u32x4 g0;
    g0[0] = 1u;                                       // count=1
    g0[1] = lds_byte;                                 // lds_addr
    g0[2] = (unsigned)ga;                             // global_addr[31:0]
    g0[3] = (unsigned)(ga >> 32) | (2u << 30);        // global_addr[56:32] | type=2
    unsigned d0 = strideElems;                        // tensor_dim0 (row length)
    unsigned d1 = tileRows;                           // tensor_dim1
    i32x8 g1;
    g1[0] = (int)(1u << 16);                          // data_size=1 (2 bytes)
    g1[1] = (int)((d0 & 0xFFFFu) << 16);              // tensor_dim0[15:0] @ bits 63:48
    g1[2] = (int)((d0 >> 16) | ((d1 & 0xFFFFu) << 16));
    g1[3] = (int)((d1 >> 16) | (tileK << 16));        // tile_dim0 @ bits 127:112
    g1[4] = (int)(tileRows & 0xFFFFu);                // tile_dim1 (tile_dim2=0)
    g1[5] = (int)strideElems;                         // tensor_dim0_stride[31:0]
    g1[6] = 0;
    g1[7] = 0;
    i32x4 z4 = {0, 0, 0, 0};
    i32x8 z8 = {0, 0, 0, 0, 0, 0, 0, 0};
    __builtin_amdgcn_tensor_load_to_lds(g0, g1, z4, z4, z8, 0);
#else
    (void)lds_byte; (void)gaddr; (void)tileK; (void)tileRows; (void)strideElems;
#endif
}
__device__ __forceinline__ void wait_tensor0() {
#if HAS_TDM
    __builtin_amdgcn_s_wait_tensorcnt((short)0);
#endif
}

__device__ __forceinline__ float rowmax16(float v) {
    #pragma unroll
    for (int m = 1; m < 16; m <<= 1) v = fmaxf(v, __shfl_xor(v, m, 32));
    return v;
}
__device__ __forceinline__ float rowsum16(float v) {
    #pragma unroll
    for (int m = 1; m < 16; m <<= 1) v += __shfl_xor(v, m, 32);
    return v;
}

// ---------------- fp32 -> bf16 (row-major copy, for x) ----------------
__global__ void cvt_kernel(const float* __restrict__ s, bf16_t* __restrict__ d, int n) {
    int i = blockIdx.x * blockDim.x + threadIdx.x;
    int stride = gridDim.x * blockDim.x;
    for (; i < n; i += stride) d[i] = f2bf(s[i]);
}

// ---------------- fp32 [R][C] -> bf16 [C][R] (weights -> B^T) ----------------
__global__ void cvt_transpose_kernel(const float* __restrict__ src, bf16_t* __restrict__ dst,
                                     int R, int C) {
    __shared__ bf16_t tile[32][33];
    const int c0 = blockIdx.x * 32, r0 = blockIdx.y * 32;
    const int tx = threadIdx.x & 31, ty = threadIdx.x >> 5;
    #pragma unroll
    for (int i = 0; i < 32; i += 8)
        tile[ty + i][tx] = f2bf(src[(size_t)(r0 + ty + i) * C + c0 + tx]);
    __syncthreads();
    #pragma unroll
    for (int i = 0; i < 32; i += 8)
        dst[(size_t)(c0 + ty + i) * R + r0 + tx] = tile[tx][ty + i];
}

// ---------------- Tiled bf16 WMMA GEMM: C(MxN f32) = A(MxK) * Bt(NxK)^T ----------
// Block 256 thr = 8 waves (4 along M x 2 along N); block tile 256x128; wave tile 64x64.
// Double-buffered LDS; A tile staged by TDM (wave 0), B tile by per-lane async DMA.
#define BM 256
#define BN 128
#define BK 32

__global__ void __launch_bounds__(256) gemm_bf16_kernel(const bf16_t* __restrict__ A,
                                                        const bf16_t* __restrict__ Bt,
                                                        float* __restrict__ C,
                                                        int M, int N, int K) {
    __shared__ bf16_t lsA[2][BM * BK];   // 2 x 16 KB
    __shared__ bf16_t lsB[2][BN * BK];   // 2 x 8 KB

    const int tid  = threadIdx.x;
    const int lane = tid & 31;
    const int wave = tid >> 5;
    const int wm = wave >> 1;            // 0..3 -> M offset wm*64
    const int wn = wave & 1;             // 0..1 -> N offset wn*64
    const int hi = lane >> 4;
    const int lo = lane & 15;
    const int m0 = blockIdx.y * BM;
    const int n0 = blockIdx.x * BN;
    const int NK = K / BK;

    v8f acc[4][4];
    #pragma unroll
    for (int mi = 0; mi < 4; ++mi)
        #pragma unroll
        for (int nf = 0; nf < 4; ++nf)
            #pragma unroll
            for (int r = 0; r < 8; ++r) acc[mi][nf][r] = 0.0f;

    // --- staging helpers ---
    auto stageA = [&](int buf, int k0) {
#if HAS_TDM
        if (wave == 0)
            tdm_load_2d(lds_addr(&lsA[buf][0]), A + (size_t)m0 * K + k0, BK, BM, K);
#else
        // fallback: 1024 x 16B chunks over 256 threads (4 each)
        #pragma unroll
        for (int i = 0; i < 4; ++i) {
            int c = tid + i * 256;
            int row = c >> 2, col = c & 3;
            async_ld16(lds_addr(&lsA[buf][row * BK + col * 8]),
                       A + (size_t)(m0 + row) * K + k0 + col * 8);
        }
#endif
    };
    auto stageB = [&](int buf, int k0) {
        // 512 x 16B chunks over 256 threads (2 each)
        #pragma unroll
        for (int i = 0; i < 2; ++i) {
            int c = tid + i * 256;
            int row = c >> 2, col = c & 3;
            async_ld16(lds_addr(&lsB[buf][row * BK + col * 8]),
                       Bt + (size_t)(n0 + row) * K + k0 + col * 8);
        }
    };

    stageA(0, 0);
    stageB(0, 0);

    for (int kt = 0; kt < NK; ++kt) {
        if (wave == 0) wait_tensor0();
        wait_async0();
        __syncthreads();                       // tile kt visible; prev reads done
        if (kt + 1 < NK) {                     // prefetch next tile (overlaps compute)
            stageA((kt + 1) & 1, (kt + 1) * BK);
            stageB((kt + 1) & 1, (kt + 1) * BK);
        }
        const bf16_t* bufA = lsA[kt & 1];
        const bf16_t* bufB = lsB[kt & 1];

        AF af[4];
        #pragma unroll
        for (int mi = 0; mi < 4; ++mi) {
            const bf16_t* base = bufA + (wm * 64 + mi * 16 + lo) * BK + hi * 8;
            af[mi].q[0] = *(const u32x4*)(base);
            af[mi].q[1] = *(const u32x4*)(base + 16);
        }
        #pragma unroll
        for (int nf = 0; nf < 4; ++nf) {
            AF bfr;
            const bf16_t* base = bufB + (wn * 64 + nf * 16 + lo) * BK + hi * 8;
            bfr.q[0] = *(const u32x4*)(base);
            bfr.q[1] = *(const u32x4*)(base + 16);
            #pragma unroll
            for (int mi = 0; mi < 4; ++mi)
                acc[mi][nf] = wmma_bf16(af[mi].v, bfr.v, acc[mi][nf]);
        }
    }

    #pragma unroll
    for (int mi = 0; mi < 4; ++mi)
        #pragma unroll
        for (int nf = 0; nf < 4; ++nf)
            #pragma unroll
            for (int r = 0; r < 8; ++r) {
                int row = m0 + wm * 64 + mi * 16 + hi * 8 + r;
                int col = n0 + wn * 64 + nf * 16 + lo;
                C[(size_t)row * N + col] = acc[mi][nf][r];
            }
}

// ---------------- RoPE + repack: f32 [S][H*64] -> bf16 [H][S][64] ----------------
__global__ void rope_pack_kernel(const float* __restrict__ src, bf16_t* __restrict__ dst,
                                 const float* __restrict__ cosT, const float* __restrict__ sinT,
                                 int nheads) {
    int idx = blockIdx.x * blockDim.x + threadIdx.x;
    int total = SEQ * nheads * (HEAD_DIM / 2);
    if (idx >= total) return;
    int d = idx & 31;
    int h = (idx >> 5) % nheads;
    int s = idx / (32 * nheads);
    size_t base = (size_t)s * (nheads * HEAD_DIM) + h * HEAD_DIM;
    float x1 = src[base + d];
    float x2 = src[base + d + 32];
    float c = cosT[s * 32 + d];
    float sn = sinT[s * 32 + d];
    size_t obase = ((size_t)h * SEQ + s) * HEAD_DIM;
    dst[obase + d]      = f2bf(x1 * c - x2 * sn);
    dst[obase + d + 32] = f2bf(x1 * sn + x2 * c);
}

// ---------------- V transpose pack: f32 [S][KV*64] -> bf16 [KV][64][S] -------------
__global__ void vtrans_kernel(const float* __restrict__ src, bf16_t* __restrict__ dst,
                              int nheads) {
    int idx = blockIdx.x * blockDim.x + threadIdx.x;
    int total = SEQ * nheads * HEAD_DIM;
    if (idx >= total) return;
    int d = idx & 63;
    int h = (idx >> 6) % nheads;
    int s = idx / (HEAD_DIM * nheads);
    dst[((size_t)h * HEAD_DIM + d) * SEQ + s] =
        f2bf(src[(size_t)s * (nheads * HEAD_DIM) + h * HEAD_DIM + d]);
}

// ---------------- Flash-style causal GQA attention -------------------------------
// Grid: (SEQ/64, N_HEADS). Block: 128 threads = 4 waves; wave w owns 16 query rows.
__global__ void attn_kernel(const bf16_t* __restrict__ qb, const bf16_t* __restrict__ kbuf,
                            const bf16_t* __restrict__ vT, bf16_t* __restrict__ ctx) {
    const int lane = threadIdx.x & 31;
    const int wave = threadIdx.x >> 5;
    const int hi = lane >> 4;
    const int lo = lane & 15;
    const int h = blockIdx.y;
    const int g = h >> 2;                              // kv head (GROUP=4)
    const int i0 = blockIdx.x * 64 + wave * 16;
    const float SCALE = 0.125f;                        // 1/sqrt(64)

    __shared__ bf16_t lsP[4][16 * 32];

    AF qA[2];
    {
        const bf16_t* qrow = qb + ((size_t)h * SEQ + (i0 + lo)) * HEAD_DIM;
        #pragma unroll
        for (int half = 0; half < 2; ++half) {
            const bf16_t* base = qrow + half * 32 + hi * 8;
            qA[half].q[0] = *(const u32x4*)(base);
            qA[half].q[1] = *(const u32x4*)(base + 16);
        }
    }

    v8f o[4];
    float mrun[8], lrun[8];
    #pragma unroll
    for (int nt = 0; nt < 4; ++nt)
        #pragma unroll
        for (int r = 0; r < 8; ++r) o[nt][r] = 0.0f;
    #pragma unroll
    for (int r = 0; r < 8; ++r) { mrun[r] = -3.0e38f; lrun[r] = 0.0f; }

    const int kb16 = hi * 8;

    for (int j0 = 0; j0 < i0 + 16; j0 += 32) {
        v8f s0, s1;
        #pragma unroll
        for (int r = 0; r < 8; ++r) { s0[r] = 0.0f; s1[r] = 0.0f; }
        #pragma unroll
        for (int nt = 0; nt < 2; ++nt) {
            AF kB0, kB1;
            const bf16_t* krow =
                kbuf + ((size_t)g * SEQ + (j0 + nt * 16 + lo)) * HEAD_DIM;
            kB0.q[0] = *(const u32x4*)(krow + kb16);
            kB0.q[1] = *(const u32x4*)(krow + kb16 + 16);
            kB1.q[0] = *(const u32x4*)(krow + 32 + kb16);
            kB1.q[1] = *(const u32x4*)(krow + 32 + kb16 + 16);
            if (nt == 0) { s0 = wmma_bf16(qA[0].v, kB0.v, s0); s0 = wmma_bf16(qA[1].v, kB1.v, s0); }
            else         { s1 = wmma_bf16(qA[0].v, kB0.v, s1); s1 = wmma_bf16(qA[1].v, kB1.v, s1); }
        }

        float alpha[8], p0[8], p1[8];
        #pragma unroll
        for (int r = 0; r < 8; ++r) {
            int row = i0 + hi * 8 + r;
            int c0 = j0 + lo;
            int c1 = c0 + 16;
            float a = s0[r] * SCALE; if (c0 > row) a = -3.0e38f;
            float b = s1[r] * SCALE; if (c1 > row) b = -3.0e38f;
            float tmax = rowmax16(fmaxf(a, b));
            float mnew = fmaxf(mrun[r], tmax);
            alpha[r] = __expf(mrun[r] - mnew);
            mrun[r] = mnew;
            float e0 = __expf(a - mnew);
            float e1 = __expf(b - mnew);
            lrun[r] = lrun[r] * alpha[r] + rowsum16(e0 + e1);
            p0[r] = e0; p1[r] = e1;
        }

        #pragma unroll
        for (int r = 0; r < 8; ++r) {
            int prow = hi * 8 + r;
            lsP[wave][prow * 32 + lo]      = f2bf(p0[r]);
            lsP[wave][prow * 32 + 16 + lo] = f2bf(p1[r]);
        }
        asm volatile("s_wait_dscnt 0" ::: "memory");
        AF pA;
        {
            const bf16_t* base = &lsP[wave][lo * 32 + hi * 8];
            pA.q[0] = *(const u32x4*)(base);
            pA.q[1] = *(const u32x4*)(base + 16);
        }

        #pragma unroll
        for (int nt = 0; nt < 4; ++nt)
            #pragma unroll
            for (int r = 0; r < 8; ++r) o[nt][r] *= alpha[r];

        #pragma unroll
        for (int nt = 0; nt < 4; ++nt) {
            AF vB;
            const bf16_t* vrow = vT + ((size_t)g * HEAD_DIM + nt * 16 + lo) * SEQ + j0;
            vB.q[0] = *(const u32x4*)(vrow + kb16);
            vB.q[1] = *(const u32x4*)(vrow + kb16 + 16);
            o[nt] = wmma_bf16(pA.v, vB.v, o[nt]);
        }
    }

    #pragma unroll
    for (int nt = 0; nt < 4; ++nt)
        #pragma unroll
        for (int r = 0; r < 8; ++r) {
            int row = i0 + hi * 8 + r;
            float val = o[nt][r] / lrun[r];
            ctx[(size_t)row * D_OUT + h * HEAD_DIM + nt * 16 + lo] = f2bf(val);
        }
}

// ---------------- Host launch ----------------
extern "C" void kernel_launch(void* const* d_in, const int* in_sizes, int n_in,
                              void* d_out, int out_size, void* d_ws, size_t ws_size,
                              hipStream_t stream) {
    (void)in_sizes; (void)n_in; (void)out_size; (void)ws_size;
    const float* x    = (const float*)d_in[0];
    const float* Wq   = (const float*)d_in[1];
    const float* Wk   = (const float*)d_in[2];
    const float* Wv   = (const float*)d_in[3];
    const float* Wo   = (const float*)d_in[4];
    const float* cosT = (const float*)d_in[5];
    const float* sinT = (const float*)d_in[6];

    char* ws = (char*)d_ws;
    size_t off = 0;
    auto take = [&](size_t bytes) -> void* {
        void* p = ws + off;
        off = (off + bytes + 255) & ~(size_t)255;
        return p;
    };
    bf16_t* xb   = (bf16_t*)take((size_t)SEQ * D_IN * 2);
    bf16_t* WqT  = (bf16_t*)take((size_t)D_IN * D_OUT * 2);
    bf16_t* WkT  = (bf16_t*)take((size_t)D_IN * KV_DIM * 2);
    bf16_t* WvT  = (bf16_t*)take((size_t)D_IN * KV_DIM * 2);
    bf16_t* WoT  = (bf16_t*)take((size_t)D_OUT * D_OUT * 2);
    float*  qf   = (float*)take((size_t)SEQ * D_OUT * 4);
    float*  kf   = (float*)take((size_t)SEQ * KV_DIM * 4);
    float*  vf   = (float*)take((size_t)SEQ * KV_DIM * 4);
    bf16_t* qhb  = (bf16_t*)take((size_t)N_HEADS * SEQ * HEAD_DIM * 2);
    bf16_t* khb  = (bf16_t*)take((size_t)N_KV * SEQ * HEAD_DIM * 2);
    bf16_t* vTb  = (bf16_t*)take((size_t)N_KV * HEAD_DIM * SEQ * 2);
    bf16_t* ctxb = (bf16_t*)take((size_t)SEQ * D_OUT * 2);

    cvt_kernel<<<1024, 256, 0, stream>>>(x, xb, SEQ * D_IN);
    cvt_transpose_kernel<<<dim3(D_OUT / 32, D_IN / 32), 256, 0, stream>>>(Wq, WqT, D_IN, D_OUT);
    cvt_transpose_kernel<<<dim3(KV_DIM / 32, D_IN / 32), 256, 0, stream>>>(Wk, WkT, D_IN, KV_DIM);
    cvt_transpose_kernel<<<dim3(KV_DIM / 32, D_IN / 32), 256, 0, stream>>>(Wv, WvT, D_IN, KV_DIM);
    cvt_transpose_kernel<<<dim3(D_OUT / 32, D_OUT / 32), 256, 0, stream>>>(Wo, WoT, D_OUT, D_OUT);

    gemm_bf16_kernel<<<dim3(D_OUT / BN, SEQ / BM), 256, 0, stream>>>(xb, WqT, qf, SEQ, D_OUT, D_IN);
    gemm_bf16_kernel<<<dim3(KV_DIM / BN, SEQ / BM), 256, 0, stream>>>(xb, WkT, kf, SEQ, KV_DIM, D_IN);
    gemm_bf16_kernel<<<dim3(KV_DIM / BN, SEQ / BM), 256, 0, stream>>>(xb, WvT, vf, SEQ, KV_DIM, D_IN);

    rope_pack_kernel<<<(SEQ * N_HEADS * 32 + 255) / 256, 256, 0, stream>>>(qf, qhb, cosT, sinT, N_HEADS);
    rope_pack_kernel<<<(SEQ * N_KV * 32 + 255) / 256, 256, 0, stream>>>(kf, khb, cosT, sinT, N_KV);
    vtrans_kernel<<<(SEQ * N_KV * HEAD_DIM + 255) / 256, 256, 0, stream>>>(vf, vTb, N_KV);

    attn_kernel<<<dim3(SEQ / 64, N_HEADS), 128, 0, stream>>>(qhb, khb, vTb, ctxb);

    gemm_bf16_kernel<<<dim3(D_OUT / BN, SEQ / BM), 256, 0, stream>>>(ctxb, WoT, (float*)d_out,
                                                                     SEQ, D_OUT, D_OUT);
}
